// NaiveFourierKANLayer_20401094656060
// MI455X (gfx1250) — matrix operand order repaired and checked by reading, not compiled, not run on hardware
//
#include <hip/hip_runtime.h>

// FourierKAN: y[n,o] = sum_{i,g} cos(x[n,i]*(g+1))*C0[o,i,g] + sin(...)*C1[o,i,g] + bias[o]
// GEMM view: A = trig features [N x 2*I*G] computed in registers, B = coeffs
// pre-transposed to K-major bf16 in d_ws, f32 accumulation via v_wmma_f32_16x16x32_bf16.

#define N_ROWS  4096
#define IN_DIM  256
#define GRID_G  64
#define OUT_DIM 512
#define KC      (IN_DIM * GRID_G)   // 16384 K per trig branch

#define USE_ASYNC_LDS 1             // global_load_async_to_lds_b128 staging path

typedef __attribute__((ext_vector_type(16))) __bf16 v16bf;
typedef __attribute__((ext_vector_type(8)))  float  v8f;

// ---------------------------------------------------------------------------
// Prep: transpose + convert coeffs [2][O][K] f32 -> [2][K][O] bf16 (in d_ws).
// ---------------------------------------------------------------------------
__global__ __launch_bounds__(256) void fkan_prep(const float* __restrict__ cf,
                                                 __bf16* __restrict__ B)
{
    __shared__ float tile[32][33];
    const int c  = blockIdx.z;
    const int kt = blockIdx.x;
    const int ot = blockIdx.y;
    const int tx = threadIdx.x;
    const int ty = threadIdx.y;

#pragma unroll
    for (int j = 0; j < 4; ++j) {
        const int o = ot * 32 + ty + 8 * j;
        tile[ty + 8 * j][tx] =
            cf[((size_t)c * OUT_DIM + o) * KC + (size_t)kt * 32 + tx];
    }
    __syncthreads();
#pragma unroll
    for (int j = 0; j < 4; ++j) {
        const int k = kt * 32 + ty + 8 * j;
        B[((size_t)c * KC + k) * OUT_DIM + (size_t)ot * 32 + tx] =
            (__bf16)tile[tx][ty + 8 * j];
    }
}

// ---------------------------------------------------------------------------
// Main GEMM. Block = 256 threads = 8 waves; block tile = 128(n) x 64(o).
// K loop: t = 0..511 steps of 32 = (i = t>>1, g-block = t&1); cos & sin fused.
// B tile (2 x 32 x 64 bf16) quad-buffered in LDS, 3-deep async pipeline.
// ---------------------------------------------------------------------------
#define LDS_PITCH 80                      // 64 cols + pad (160B rows, 32B mult)
#define LDS_ROWB  (LDS_PITCH * 2)         // 160
#define PLANE_B   (32 * LDS_ROWB)         // 5120  (one trig plane)
#define BUF_B     (2 * PLANE_B)           // 10240 (one K-step tile)

__global__ __launch_bounds__(256) void fkan_gemm(const float* __restrict__ x,
                                                 const __bf16* __restrict__ B,
                                                 const float* __restrict__ bias,
                                                 float* __restrict__ out)
{
    __shared__ __align__(32) __bf16 smem[4][2][32][LDS_PITCH];

    const int tid  = threadIdx.x;
    const int lane = tid & 31;
    const int w    = tid >> 5;
    const int n0   = blockIdx.x * 128;
    const int o0   = blockIdx.y * 64;
    const int h    = lane >> 4;
    const int nrow = n0 + w * 16 + (lane & 15);

    // staging decomposition: each thread moves one 16B chunk per trig plane
    const int kk = tid >> 3;              // 0..31 (K row in tile)
    const int j8 = tid & 7;               // 0..7  (16B chunk along O)
    const unsigned ldsChunk = (unsigned)(kk * LDS_ROWB + j8 * 16);

    // Flat LDS pointer: low 32 bits are the wave-relative LDS byte address
    // (same convention ds_* / async VDST use). Deriving the asm operands from
    // this pointer also captures smem, so the asm "memory" clobber is honored.
    const char* smembase = (const char*)&smem[0][0][0][0];

    v8f acc[4] = {};

#if USE_ASYNC_LDS
    // Async memory->LDS staging: GV mode, VDST holds LDS byte address.
    auto issue = [&](int t) {
        const int buf = t & 3;
        const size_t kb = (size_t)t * 32 + kk;
        const unsigned long long g0 =
            (unsigned long long)(uintptr_t)(B + kb * OUT_DIM + o0 + j8 * 8);
        const unsigned long long g1 =
            (unsigned long long)(uintptr_t)(B + (kb + KC) * OUT_DIM + o0 + j8 * 8);
        const unsigned l0 = (unsigned)(uintptr_t)(smembase +
                                (size_t)buf * BUF_B + ldsChunk);
        const unsigned l1 = (unsigned)(uintptr_t)(smembase +
                                (size_t)buf * BUF_B + PLANE_B + ldsChunk);
        asm volatile(
            "global_load_async_to_lds_b128 %0, %1, off\n\t"
            "global_load_async_to_lds_b128 %2, %3, off"
            :: "v"(l0), "v"(g0), "v"(l1), "v"(g1)
            : "memory");
    };
#else
    int4 r0, r1;
    auto gload = [&](int t) {
        const size_t kb = (size_t)t * 32 + kk;
        r0 = *(const int4*)(B + kb * OUT_DIM + o0 + j8 * 8);
        r1 = *(const int4*)(B + (kb + KC) * OUT_DIM + o0 + j8 * 8);
    };
    auto sstore = [&](int t) {
        const int buf = t & 3;
        *(int4*)((char*)smembase + (size_t)buf * BUF_B + ldsChunk) = r0;
        *(int4*)((char*)smembase + (size_t)buf * BUF_B + PLANE_B + ldsChunk) = r1;
    };
#endif

    const int T = 2 * IN_DIM;             // 512 K-steps

#if USE_ASYNC_LDS
    issue(0);
    issue(1);
    issue(2);
    asm volatile("s_wait_asynccnt 0x4" ::: "memory");  // step 0 resident
    __syncthreads();
#else
    gload(0); sstore(0); gload(1); sstore(1);
    __syncthreads();
#endif

    for (int i = 0; i < IN_DIM; ++i) {
        const float xv = x[(size_t)nrow * IN_DIM + i];
#pragma unroll
        for (int gb = 0; gb < 2; ++gb) {
            const int t = 2 * i + gb;

            // A fragments per the 16-bit A 16x32 VGPR layout:
            // slot j < 8 -> K = 8h + j ; slot j >= 8 -> K = 16 + 8h + (j-8)
            v16bf ac, as;
#pragma unroll
            for (int j = 0; j < 16; ++j) {
                const int kkA = (j < 8) ? (8 * h + j) : (16 + 8 * h + (j - 8));
                const float ang = xv * (float)(gb * 32 + kkA + 1);
                float sv, cv;
                __sincosf(ang, &sv, &cv);
                ac[j] = (__bf16)cv;
                as[j] = (__bf16)sv;
            }

            const int buf = t & 3;
            const char* base = smembase + (size_t)buf * BUF_B;

            // Hoist all 8 B fragments (strided 16-bit LDS gathers) ahead of the WMMAs so
            // LDS latency overlaps the trig VALU work instead of serializing.
            v16bf bfrag[8];
#pragma unroll
            for (int s = 0; s < 4; ++s) {
#pragma unroll
                for (int e = 0; e < 16; ++e) {
                    const int kB = (e < 8) ? (8 * h + e) : (16 + 8 * h + (e - 8));
                    const size_t off = (size_t)kB * LDS_ROWB + (s * 16 + (lane & 15)) * 2;
                    bfrag[s][e]     = *(const __bf16*)(base + off);
                    bfrag[4 + s][e] = *(const __bf16*)(base + PLANE_B + off);
                }
            }
#pragma unroll
            for (int s = 0; s < 4; ++s)
                acc[s] = __builtin_amdgcn_wmma_f32_16x16x32_bf16(
                    false, ac, false, bfrag[s], (short)0, acc[s], false, false);
#pragma unroll
            for (int s = 0; s < 4; ++s)
                acc[s] = __builtin_amdgcn_wmma_f32_16x16x32_bf16(
                    false, as, false, bfrag[4 + s], (short)0, acc[s], false, false);

#if USE_ASYNC_LDS
            if (t + 3 < T) {
                issue(t + 3);
                // in-order async completion: <=4 outstanding => step t+1 resident
                asm volatile("s_wait_asynccnt 0x4" ::: "memory");
            } else if (t + 2 < T) {
                asm volatile("s_wait_asynccnt 0x2" ::: "memory");
            } else {
                asm volatile("s_wait_asynccnt 0x0" ::: "memory");
            }
            __syncthreads();
#else
            if (t + 2 < T) { gload(t + 2); sstore(t + 2); }
            __syncthreads();
#endif
        }
    }

    // Epilogue per the 32-bit C/D 16x16 layout: lane -> N (o), VGPR r -> M row
#pragma unroll
    for (int s = 0; s < 4; ++s) {
        const int o  = o0 + s * 16 + (lane & 15);
        const float bv = bias[o];
#pragma unroll
        for (int r = 0; r < 8; ++r) {
            const int n = n0 + w * 16 + h * 8 + r;
            out[(size_t)n * OUT_DIM + o] = acc[s][r] + bv;
        }
    }
}

// ---------------------------------------------------------------------------
extern "C" void kernel_launch(void* const* d_in, const int* in_sizes, int n_in,
                              void* d_out, int out_size, void* d_ws, size_t ws_size,
                              hipStream_t stream)
{
    const float* x    = (const float*)d_in[0];   // [4096, 256]
    const float* cf   = (const float*)d_in[1];   // [2, 512, 256, 64]
    const float* bias = (const float*)d_in[2];   // [1, 512]
    float* out        = (float*)d_out;           // [4096, 512]
    __bf16* Bws       = (__bf16*)d_ws;           // needs 32 MB scratch

    dim3 pgrid(KC / 32, OUT_DIM / 32, 2);        // (512, 16, 2)
    fkan_prep<<<pgrid, dim3(32, 8), 0, stream>>>(cf, Bws);

    dim3 ggrid(N_ROWS / 128, OUT_DIM / 64);      // (32, 8)
    fkan_gemm<<<ggrid, 256, 0, stream>>>(x, Bws, bias, out);
}
